// OmniSuperPointTransformer_24172075942043
// MI455X (gfx1250) — matrix operand order repaired and checked by reading, not compiled, not run on hardware
//
#include <hip/hip_runtime.h>
#include <hip/hip_bf16.h>

typedef __attribute__((ext_vector_type(16))) _Float16 v16h;
typedef __attribute__((ext_vector_type(8)))  float    v8f;

#define CH 32
#define LN_EPS 1e-5f
#define TILES_PER_WAVE 4
#define WAVES_PER_BLOCK 8

// ---------------------------------------------------------------------------
// Zero the workspace accumulators (graph-capture safe)
// ---------------------------------------------------------------------------
__global__ void osp_zero_ws(float* __restrict__ ws, int n) {
    int i = blockIdx.x * blockDim.x + threadIdx.x;
    if (i < n) ws[i] = 0.0f;
}

// ---------------------------------------------------------------------------
// Fused full-tile kernel: gather voxel feats + pos-MLP (Linear3->32, LN,
// ReLU, WMMA 32x32) + segment-sum scatter. Branch-free per tile; one wave32
// handles TILES_PER_WAVE consecutive 16-point tiles. 32-bit offsets only.
// ---------------------------------------------------------------------------
__global__ __launch_bounds__(256)
void osp_fused_mlp_scatter(
    const float* __restrict__ voxel_feats,   // [V, 32]
    const float* __restrict__ xyz,           // [N, 3]
    const float* __restrict__ W1,            // [3, 32]
    const float* __restrict__ b1,            // [32]
    const float* __restrict__ gamma,         // [32]
    const float* __restrict__ beta,          // [32]
    const float* __restrict__ W2,            // [32, 32]
    const float* __restrict__ b2,            // [32]
    const int*   __restrict__ p2v,           // [N]
    const int*   __restrict__ sp,            // [N]
    float* __restrict__ acc,                 // [S*32] feature sums
    float* __restrict__ acc_xyz,             // [S*3]  xyz sums
    float* __restrict__ cnt,                 // [S]    counts
    int ntiles_full)                         // npts_full / 16
{
    const int lane  = threadIdx.x & 31;
    const int wave  = threadIdx.x >> 5;
    const int tile0 = (blockIdx.x * WAVES_PER_BLOCK + wave) * TILES_PER_WAVE;
    if (tile0 >= ntiles_full) return;        // wave-uniform

    const int half  = lane >> 4;             // lane half selects K-range
    const int mlane = lane & 15;

    // ---- per-wave constants ----
    // W2 B-fragments: lane<16 holds col N=mlane, K={0..7,16..23} in halfs 0..15;
    //                 lane>=16 holds col N=mlane, K={8..15,24..31}.
    v16h bf0, bf1;
    #pragma unroll
    for (int j = 0; j < 16; ++j) {
        const int k = (j < 8) ? (8 * half + j) : (16 + 8 * half + (j - 8));
        bf0[j] = (_Float16)W2[k * CH + mlane];
        bf1[j] = (_Float16)W2[k * CH + mlane + 16];
    }
    const float bias0 = b2[mlane];
    const float bias1 = b2[mlane + 16];

    // A-fragment channel sets are two contiguous 8-blocks:
    //   lane<16 : {0..7, 16..23}   lane>=16 : {8..15, 24..31}
    // Load W1/b1/gamma/beta for them as float4s (b128 loads).
    const int cb0 = 8 * half;                // first block start
    const int cb1 = 16 + 8 * half;           // second block start
    float w1r[3][16], b1r[16], gr[16], br[16];
    #pragma unroll
    for (int k = 0; k < 3; ++k) {
        const float4 t0 = *(const float4*)(W1 + k * CH + cb0);
        const float4 t1 = *(const float4*)(W1 + k * CH + cb0 + 4);
        const float4 t2 = *(const float4*)(W1 + k * CH + cb1);
        const float4 t3 = *(const float4*)(W1 + k * CH + cb1 + 4);
        w1r[k][0]=t0.x;  w1r[k][1]=t0.y;  w1r[k][2]=t0.z;  w1r[k][3]=t0.w;
        w1r[k][4]=t1.x;  w1r[k][5]=t1.y;  w1r[k][6]=t1.z;  w1r[k][7]=t1.w;
        w1r[k][8]=t2.x;  w1r[k][9]=t2.y;  w1r[k][10]=t2.z; w1r[k][11]=t2.w;
        w1r[k][12]=t3.x; w1r[k][13]=t3.y; w1r[k][14]=t3.z; w1r[k][15]=t3.w;
    }
    {
        const float4 u0 = *(const float4*)(b1 + cb0);
        const float4 u1 = *(const float4*)(b1 + cb0 + 4);
        const float4 u2 = *(const float4*)(b1 + cb1);
        const float4 u3 = *(const float4*)(b1 + cb1 + 4);
        b1r[0]=u0.x; b1r[1]=u0.y; b1r[2]=u0.z; b1r[3]=u0.w;
        b1r[4]=u1.x; b1r[5]=u1.y; b1r[6]=u1.z; b1r[7]=u1.w;
        b1r[8]=u2.x; b1r[9]=u2.y; b1r[10]=u2.z; b1r[11]=u2.w;
        b1r[12]=u3.x; b1r[13]=u3.y; b1r[14]=u3.z; b1r[15]=u3.w;
        const float4 g0 = *(const float4*)(gamma + cb0);
        const float4 g1 = *(const float4*)(gamma + cb0 + 4);
        const float4 g2 = *(const float4*)(gamma + cb1);
        const float4 g3 = *(const float4*)(gamma + cb1 + 4);
        gr[0]=g0.x; gr[1]=g0.y; gr[2]=g0.z; gr[3]=g0.w;
        gr[4]=g1.x; gr[5]=g1.y; gr[6]=g1.z; gr[7]=g1.w;
        gr[8]=g2.x; gr[9]=g2.y; gr[10]=g2.z; gr[11]=g2.w;
        gr[12]=g3.x; gr[13]=g3.y; gr[14]=g3.z; gr[15]=g3.w;
        const float4 e0 = *(const float4*)(beta + cb0);
        const float4 e1 = *(const float4*)(beta + cb0 + 4);
        const float4 e2 = *(const float4*)(beta + cb1);
        const float4 e3 = *(const float4*)(beta + cb1 + 4);
        br[0]=e0.x; br[1]=e0.y; br[2]=e0.z; br[3]=e0.w;
        br[4]=e1.x; br[5]=e1.y; br[6]=e1.z; br[7]=e1.w;
        br[8]=e2.x; br[9]=e2.y; br[10]=e2.z; br[11]=e2.w;
        br[12]=e3.x; br[13]=e3.y; br[14]=e3.z; br[15]=e3.w;
    }

    #pragma unroll 1
    for (int t = 0; t < TILES_PER_WAVE; ++t) {
        const int tile = tile0 + t;
        if (tile >= ntiles_full) break;      // wave-uniform
        const int base = tile * 16;

        // ---- this lane's A-fragment point (row M = mlane) ----
        const int p_a = base + mlane;
        const float x0 = xyz[p_a * 3 + 0];
        const float x1 = xyz[p_a * 3 + 1];
        const float x2 = xyz[p_a * 3 + 2];
        const int sid_own = sp[p_a];         // coalesced once per point
        const int vid_own = p2v[p_a];

        // ---- h = xyz @ W1 + b1 (A-fragment channel order) + LN stats ----
        float hv[16];
        float sum = 0.f, ssq = 0.f;
        #pragma unroll
        for (int j = 0; j < 16; ++j) {
            const float v = x0 * w1r[0][j] + x1 * w1r[1][j] + x2 * w1r[2][j] + b1r[j];
            hv[j] = v; sum += v; ssq += v * v;
        }
        sum += __shfl_xor(sum, 16, 32);      // combine with partner lane
        ssq += __shfl_xor(ssq, 16, 32);
        const float mean = sum * (1.0f / 32.0f);
        const float var  = ssq * (1.0f / 32.0f) - mean * mean;
        const float rstd = rsqrtf(var + LN_EPS);

        v16h a;
        #pragma unroll
        for (int j = 0; j < 16; ++j) {
            float v = (hv[j] - mean) * rstd * gr[j] + br[j];
            v = v > 0.0f ? v : 0.0f;
            a[j] = (_Float16)v;
        }

        // ---- D = relu(LN(h)) @ W2 + b2 (C seeded with bias) ----
        v8f c0, c1;
        #pragma unroll
        for (int r = 0; r < 8; ++r) { c0[r] = bias0; c1[r] = bias1; }
        v8f d0 = __builtin_amdgcn_wmma_f32_16x16x32_f16(false, a, false, bf0,
                                                        (short)0, c0, false, false);
        v8f d1 = __builtin_amdgcn_wmma_f32_16x16x32_f16(false, a, false, bf1,
                                                        (short)0, c1, false, false);

        // ---- redistribute sp/p2v to D layout (lane owns col mlane,
        //      rows M = r + 8*half) via lane permutes; 32-bit offsets ----
        int soff[8], voff[8];
        #pragma unroll
        for (int r = 0; r < 8; ++r) {
            soff[r] = __shfl(sid_own, 8 * half + r, 32) * CH + mlane;
            voff[r] = __shfl(vid_own, 8 * half + r, 32) * CH + mlane;
        }

        // ---- branch-free gather (all loads in flight) then atomics ----
        float g0[8], g1[8];
        #pragma unroll
        for (int r = 0; r < 8; ++r) {
            g0[r] = voxel_feats[voff[r]];
            g1[r] = voxel_feats[voff[r] + 16];
        }
        #pragma unroll
        for (int r = 0; r < 8; ++r) {
            atomicAdd(&acc[soff[r]],      g0[r] + d0[r]);
            atomicAdd(&acc[soff[r] + 16], g1[r] + d1[r]);
        }
        if (half == 0) {                     // one lane per point
            atomicAdd(&cnt[sid_own], 1.0f);
            atomicAdd(&acc_xyz[sid_own * 3 + 0], x0);
            atomicAdd(&acc_xyz[sid_own * 3 + 1], x1);
            atomicAdd(&acc_xyz[sid_own * 3 + 2], x2);
        }
    }
}

// ---------------------------------------------------------------------------
// Scalar tail for the <16 remainder points (not hit when npts % 16 == 0)
// ---------------------------------------------------------------------------
__global__ void osp_tail(
    const float* __restrict__ voxel_feats, const float* __restrict__ xyz,
    const float* __restrict__ W1, const float* __restrict__ b1,
    const float* __restrict__ gamma, const float* __restrict__ beta,
    const float* __restrict__ W2, const float* __restrict__ b2,
    const int* __restrict__ p2v, const int* __restrict__ sp,
    float* __restrict__ acc, float* __restrict__ acc_xyz, float* __restrict__ cnt,
    int pstart, int npts)
{
    const int p = pstart + blockIdx.x * blockDim.x + threadIdx.x;
    if (p >= npts) return;
    const float x0 = xyz[p*3+0], x1 = xyz[p*3+1], x2 = xyz[p*3+2];
    float h[CH]; float s = 0.f, sq = 0.f;
    for (int c = 0; c < CH; ++c) {
        float v = x0*W1[c] + x1*W1[CH+c] + x2*W1[2*CH+c] + b1[c];
        h[c] = v; s += v; sq += v*v;
    }
    const float mean = s * (1.0f/CH);
    const float var  = sq * (1.0f/CH) - mean*mean;
    const float rstd = rsqrtf(var + LN_EPS);
    float rl[CH];
    for (int c = 0; c < CH; ++c) {
        float v = (h[c]-mean)*rstd*gamma[c] + beta[c];
        rl[c] = v > 0.f ? v : 0.f;
    }
    const int sid = sp[p], vid = p2v[p];
    for (int c = 0; c < CH; ++c) {
        float xp = b2[c];
        for (int k = 0; k < CH; ++k) xp += rl[k] * W2[k*CH + c];
        atomicAdd(&acc[sid*CH + c], voxel_feats[vid*CH + c] + xp);
    }
    atomicAdd(&cnt[sid], 1.0f);
    atomicAdd(&acc_xyz[sid*3+0], x0);
    atomicAdd(&acc_xyz[sid*3+1], x1);
    atomicAdd(&acc_xyz[sid*3+2], x2);
}

// ---------------------------------------------------------------------------
// Finalize: divide sums by max(count,1); write (x_sp [S,32], sp_xyz [S,3]) flat
// ---------------------------------------------------------------------------
__global__ void osp_finalize(const float* __restrict__ acc,
                             const float* __restrict__ acc_xyz,
                             const float* __restrict__ cnt,
                             float* __restrict__ out, int S)
{
    const int i   = blockIdx.x * blockDim.x + threadIdx.x;
    const int n32 = S * CH;
    if (i < n32) {
        const int s = i >> 5;
        float c = cnt[s]; c = c > 1.0f ? c : 1.0f;
        out[i] = acc[i] / c;
    } else if (i < n32 + S * 3) {
        const int j = i - n32;
        const int s = j / 3;
        float c = cnt[s]; c = c > 1.0f ? c : 1.0f;
        out[n32 + j] = acc_xyz[j] / c;
    }
}

// ---------------------------------------------------------------------------
extern "C" void kernel_launch(void* const* d_in, const int* in_sizes, int n_in,
                              void* d_out, int out_size, void* d_ws, size_t ws_size,
                              hipStream_t stream)
{
    const float* voxel_feats = (const float*)d_in[0];
    const float* xyz         = (const float*)d_in[1];
    const float* W1          = (const float*)d_in[2];
    const float* b1          = (const float*)d_in[3];
    const float* gamma       = (const float*)d_in[4];
    const float* beta        = (const float*)d_in[5];
    const float* W2          = (const float*)d_in[6];
    const float* b2          = (const float*)d_in[7];
    const int*   p2v         = (const int*)d_in[8];
    const int*   sp          = (const int*)d_in[9];
    float* out = (float*)d_out;

    const int npts = in_sizes[1] / 3;        // xyz is [N,3]
    const int S    = out_size / (CH + 3);    // out = S*32 + S*3 floats

    float* acc     = (float*)d_ws;           // [S*32]
    float* acc_xyz = acc + (size_t)S * CH;   // [S*3]
    float* cnt     = acc_xyz + (size_t)S * 3;// [S]

    // 1) zero accumulators
    {
        const int n = S * (CH + 3 + 1);
        osp_zero_ws<<<(n + 255) / 256, 256, 0, stream>>>((float*)d_ws, n);
    }
    // 2) fused gather + MLP (WMMA) + scatter over all full 16-point tiles
    const int ntiles_full = npts / 16;
    if (ntiles_full > 0) {
        const int waves  = (ntiles_full + TILES_PER_WAVE - 1) / TILES_PER_WAVE;
        const int blocks = (waves + WAVES_PER_BLOCK - 1) / WAVES_PER_BLOCK;
        osp_fused_mlp_scatter<<<blocks, 256, 0, stream>>>(
            voxel_feats, xyz, W1, b1, gamma, beta, W2, b2, p2v, sp,
            acc, acc_xyz, cnt, ntiles_full);
    }
    // 2b) scalar tail for remainder points (deterministic host-side branch)
    const int npts_full = ntiles_full * 16;
    if (npts_full < npts) {
        osp_tail<<<1, 16, 0, stream>>>(
            voxel_feats, xyz, W1, b1, gamma, beta, W2, b2, p2v, sp,
            acc, acc_xyz, cnt, npts_full, npts);
    }
    // 3) finalize means -> d_out
    {
        const int n = S * (CH + 3);
        osp_finalize<<<(n + 255) / 256, 256, 0, stream>>>(acc, acc_xyz, cnt, out, S);
    }
}
